// Attention_51135880626747
// MI455X (gfx1250) — matrix operand order repaired and checked by reading, not compiled, not run on hardware
//
#include <hip/hip_runtime.h>

// ---------------------------------------------------------------------------
// Causal attention (B=8, T=2048, C=1024), bf16 WMMA compute, f32 accumulate.
//   1) qkv_kernel    : Q/K/V = x @ W^T + b      (reg-double-buffered bf16 WMMA)
//   2) scores_kernel : S = (K @ Q^T)/32, causal (async-to-LDS ping-pong WMMA)
//   3) softmax_kernel: P = softmax_row(S) -> bf16, zero above diagonal
//   4) av_kernel     : out = P @ V              (async A, staged-transpose V)
// ---------------------------------------------------------------------------

typedef __bf16 bf16_t;
typedef bf16_t bf16x4 __attribute__((ext_vector_type(4)));
typedef bf16_t bf16x8 __attribute__((ext_vector_type(8)));
typedef bf16_t v16bf  __attribute__((ext_vector_type(16)));
typedef float  v8f    __attribute__((ext_vector_type(8)));

#define T_DIM 2048
#define C_DIM 1024
#define LDSK  40   // 32 k-elems + 8 pad: 80B rows -> 16B aligned, bank-conflict free

#define USE_ASYNC 1   // CDNA5 global_load_async_to_lds_b128 path (inline asm)

#if defined(__has_builtin)
#if __has_builtin(__builtin_amdgcn_s_wait_asynccnt)
#define WAIT_ASYNC() __builtin_amdgcn_s_wait_asynccnt(0)
#else
#define WAIT_ASYNC() asm volatile("s_wait_asynccnt 0x0" ::: "memory")
#endif
#else
#define WAIT_ASYNC() asm volatile("s_wait_asynccnt 0x0" ::: "memory")
#endif

__device__ __forceinline__ v8f zero_v8f() {
  v8f z;
#pragma unroll
  for (int i = 0; i < 8; ++i) z[i] = 0.0f;
  return z;
}

// A fragment: 16x32 (MxK) bf16. lanes 0-15 -> M, half-wave selects K-half.
__device__ __forceinline__ v16bf frag_a(const bf16_t* lds) {
  const int lane = threadIdx.x & 31;
  const bf16_t* p = lds + (lane & 15) * LDSK + ((lane >> 4) << 3);
  v16bf a;
#pragma unroll
  for (int j = 0; j < 8; ++j) { a[j] = p[j]; a[8 + j] = p[16 + j]; }
  return a;
}

// B fragment: 32x16 (KxN) bf16 staged as [n][k] rows; k = 8h+j and 16+8h+j per lane.
__device__ __forceinline__ v16bf frag_b(const bf16_t* lds) {
  const int lane = threadIdx.x & 31;
  const bf16_t* p = lds + (lane & 15) * LDSK + ((lane >> 4) << 3);
  v16bf b;
#pragma unroll
  for (int j = 0; j < 8; ++j) { b[j] = p[j]; b[8 + j] = p[16 + j]; }
  return b;
}

// 8 waves: 4 (m) x 2 (n); each wave owns 32x64 = 2x4 WMMA tiles.
__device__ __forceinline__ void mma_tiles(const bf16_t* ldsA, const bf16_t* ldsB,
                                          int wm, int wn, v8f acc[2][4]) {
  v16bf a0 = frag_a(ldsA + (wm * 32 + 0)  * LDSK);
  v16bf a1 = frag_a(ldsA + (wm * 32 + 16) * LDSK);
#pragma unroll
  for (int j = 0; j < 4; ++j) {
    v16bf b = frag_b(ldsB + (wn * 64 + j * 16) * LDSK);
    acc[0][j] = __builtin_amdgcn_wmma_f32_16x16x32_bf16(
        false, a0, false, b, (short)0, acc[0][j], false, false);
    acc[1][j] = __builtin_amdgcn_wmma_f32_16x16x32_bf16(
        false, a1, false, b, (short)0, acc[1][j], false, false);
  }
}

// ---- staging helpers (fully unrolled, constant trip counts) ---------------
// 128x32 fp32 tile -> 4 float4 per thread
__device__ __forceinline__ void gload_f32_tile(const float* g, int k0, float4 r[4]) {
#pragma unroll
  for (int u = 0; u < 4; ++u) {
    int i = (int)threadIdx.x + u * 256;
    int row = i >> 3, seg = i & 7;
    r[u] = *(const float4*)(g + (size_t)row * C_DIM + k0 + seg * 4);
  }
}
__device__ __forceinline__ void store_f32_tile(bf16_t* lds, const float4 r[4]) {
#pragma unroll
  for (int u = 0; u < 4; ++u) {
    int i = (int)threadIdx.x + u * 256;
    int row = i >> 3, seg = i & 7;
    bf16x4 h = { (bf16_t)r[u].x, (bf16_t)r[u].y, (bf16_t)r[u].z, (bf16_t)r[u].w };
    *(bf16x4*)(lds + row * LDSK + seg * 4) = h;
  }
}
// 128x32 bf16 tile -> 2 bf16x8 per thread
__device__ __forceinline__ void gload_bf16_tile(const bf16_t* g, long ldg, int k0,
                                                bf16x8 r[2]) {
#pragma unroll
  for (int u = 0; u < 2; ++u) {
    int i = (int)threadIdx.x + u * 256;
    int row = i >> 2, seg = i & 3;
    r[u] = *(const bf16x8*)(g + (size_t)row * ldg + k0 + seg * 8);
  }
}
__device__ __forceinline__ void store_bf16_tile(bf16_t* lds, const bf16x8 r[2]) {
#pragma unroll
  for (int u = 0; u < 2; ++u) {
    int i = (int)threadIdx.x + u * 256;
    int row = i >> 2, seg = i & 3;
    *(bf16x8*)(lds + row * LDSK + seg * 8) = r[u];
  }
}
// V tile 32(s) x 128(c) -> 2 bf16x8 per thread (g already offset by c0)
__device__ __forceinline__ void gload_v_tile(const bf16_t* g, int s0, bf16x8 r[2]) {
#pragma unroll
  for (int u = 0; u < 2; ++u) {
    int i = (int)threadIdx.x + u * 256;
    int s = i >> 4, seg = i & 15;
    r[u] = *(const bf16x8*)(g + (size_t)(s0 + s) * C_DIM + seg * 8);
  }
}
// scatter-store V transposed into LDS [c][s]
__device__ __forceinline__ void store_vT_tile(bf16_t* lds, const bf16x8 r[2]) {
#pragma unroll
  for (int u = 0; u < 2; ++u) {
    int i = (int)threadIdx.x + u * 256;
    int s = i >> 4, seg = i & 15;
#pragma unroll
    for (int j = 0; j < 8; ++j) lds[(seg * 8 + j) * LDSK + s] = r[u][j];
  }
}

#if USE_ASYNC
// One 16B per-lane async copy HBM -> LDS (ASYNCcnt tracked, no VGPR data).
// GV mode: vdst = LDS byte offset (generic LDS ptr low 32 bits), vaddr = 64b.
__device__ __forceinline__ void async_load_b128(const bf16_t* g, bf16_t* lds) {
  unsigned lds_off = (unsigned)(uintptr_t)lds;
  unsigned long long ga = (unsigned long long)(uintptr_t)g;
  asm volatile("global_load_async_to_lds_b128 %0, %1, off"
               :: "v"(lds_off), "v"(ga) : "memory");
}
// 128x32 bf16 tile, HBM -> LDS
__device__ __forceinline__ void async_bf16_tile(const bf16_t* g, long ldg,
                                                bf16_t* lds) {
#pragma unroll
  for (int u = 0; u < 2; ++u) {
    int i = (int)threadIdx.x + u * 256;
    int row = i >> 2, seg = i & 3;
    async_load_b128(g + (size_t)row * ldg + seg * 8,
                    lds + row * LDSK + seg * 8);
  }
}
#endif

// ---------------------------------------------------------------------------
// Kernel 1: QKV projections. out[m,n] = sum_k x[m,k]*w[n,k] + bias[n]
// fp32 source requires VALU convert -> register double-buffer pipeline.
// Prefetch loads issued AFTER the visibility barrier so __syncthreads'
// s_wait_loadcnt_dscnt does not stall on them; they fly under the WMMAs.
// ---------------------------------------------------------------------------
__global__ __launch_bounds__(256) void qkv_kernel(
    const float* __restrict__ x,
    const float* __restrict__ wq, const float* __restrict__ bq,
    const float* __restrict__ wk, const float* __restrict__ bk,
    const float* __restrict__ wv, const float* __restrict__ bv,
    bf16_t* __restrict__ Q, bf16_t* __restrict__ K, bf16_t* __restrict__ V)
{
  __shared__ bf16_t ldsA[128 * LDSK];
  __shared__ bf16_t ldsB[128 * LDSK];

  const float* w; const float* bias; bf16_t* out;
  if (blockIdx.z == 0)      { w = wq; bias = bq; out = Q; }
  else if (blockIdx.z == 1) { w = wk; bias = bk; out = K; }
  else                      { w = wv; bias = bv; out = V; }

  const int m0 = blockIdx.x * 128;
  const int n0 = blockIdx.y * 128;
  const int tid  = threadIdx.x;
  const int wave = tid >> 5, wm = wave >> 1, wn = wave & 1;
  const float* gA = x + (size_t)m0 * C_DIM;
  const float* gB = w + (size_t)n0 * C_DIM;

  v8f acc[2][4];
#pragma unroll
  for (int i = 0; i < 2; ++i)
#pragma unroll
    for (int j = 0; j < 4; ++j) acc[i][j] = zero_v8f();

  float4 ra[4], rb[4];
  gload_f32_tile(gA, 0, ra);
  gload_f32_tile(gB, 0, rb);

  for (int kt = 0; kt < 32; ++kt) {
    __syncthreads();                      // previous iter's LDS reads done
    store_f32_tile(ldsA, ra);
    store_f32_tile(ldsB, rb);
    __syncthreads();                      // stores visible
    if (kt + 1 < 32) {                    // issue AFTER barrier: hides under mma
      gload_f32_tile(gA, (kt + 1) * 32, ra);
      gload_f32_tile(gB, (kt + 1) * 32, rb);
    }
    mma_tiles(ldsA, ldsB, wm, wn, acc);
  }

  const int lane  = tid & 31;
  const int mBase = m0 + wm * 32 + ((lane >> 4) << 3);
  const int nBase = n0 + wn * 64 + (lane & 15);
#pragma unroll
  for (int i = 0; i < 2; ++i)
#pragma unroll
    for (int j = 0; j < 4; ++j) {
      int n = nBase + j * 16;
      float bb = bias[n];
#pragma unroll
      for (int r = 0; r < 8; ++r)
        out[(size_t)(mBase + i * 16 + r) * C_DIM + n] = (bf16_t)(acc[i][j][r] + bb);
    }
}

// ---------------------------------------------------------------------------
// Kernel 2: S[b,t,s] = (K[b,t,:].Q[b,s,:])/32, causal. Upper tiles early-exit.
// ---------------------------------------------------------------------------
__global__ __launch_bounds__(256) void scores_kernel(
    const bf16_t* __restrict__ Kb, const bf16_t* __restrict__ Qb,
    float* __restrict__ S)
{
  if (blockIdx.y > blockIdx.x) return;

  const int b  = blockIdx.z;
  const int t0 = blockIdx.x * 128;
  const int s0 = blockIdx.y * 128;
  const bf16_t* Kp = Kb + (size_t)b * T_DIM * C_DIM + (size_t)t0 * C_DIM;
  const bf16_t* Qp = Qb + (size_t)b * T_DIM * C_DIM + (size_t)s0 * C_DIM;

  const int tid  = threadIdx.x;
  const int wave = tid >> 5, wm = wave >> 1, wn = wave & 1;

  v8f acc[2][4];
#pragma unroll
  for (int i = 0; i < 2; ++i)
#pragma unroll
    for (int j = 0; j < 4; ++j) acc[i][j] = zero_v8f();

#if USE_ASYNC
  __shared__ bf16_t ldsA[2][128 * LDSK];
  __shared__ bf16_t ldsB[2][128 * LDSK];
  async_bf16_tile(Kp, C_DIM, ldsA[0]);
  async_bf16_tile(Qp, C_DIM, ldsB[0]);
  for (int kt = 0; kt < 32; ++kt) {
    const int cur = kt & 1;
    WAIT_ASYNC();                         // own tile(kt) copies landed
    __syncthreads();                      // everyone's landed; spare buffer free
    if (kt + 1 < 32) {
      async_bf16_tile(Kp + (kt + 1) * 32, C_DIM, ldsA[cur ^ 1]);
      async_bf16_tile(Qp + (kt + 1) * 32, C_DIM, ldsB[cur ^ 1]);
    }
    mma_tiles(ldsA[cur], ldsB[cur], wm, wn, acc);
  }
#else
  __shared__ bf16_t ldsA[128 * LDSK];
  __shared__ bf16_t ldsB[128 * LDSK];
  bf16x8 ra[2], rb[2];
  gload_bf16_tile(Kp, C_DIM, 0, ra);
  gload_bf16_tile(Qp, C_DIM, 0, rb);
  for (int kt = 0; kt < 32; ++kt) {
    __syncthreads();
    store_bf16_tile(ldsA, ra);
    store_bf16_tile(ldsB, rb);
    __syncthreads();
    if (kt + 1 < 32) {
      gload_bf16_tile(Kp, C_DIM, (kt + 1) * 32, ra);
      gload_bf16_tile(Qp, C_DIM, (kt + 1) * 32, rb);
    }
    mma_tiles(ldsA, ldsB, wm, wn, acc);
  }
#endif

  const int lane  = tid & 31;
  const int tBase = t0 + wm * 32 + ((lane >> 4) << 3);
  const int sBase = s0 + wn * 64 + (lane & 15);
#pragma unroll
  for (int i = 0; i < 2; ++i)
#pragma unroll
    for (int j = 0; j < 4; ++j) {
      int s = sBase + j * 16;
#pragma unroll
      for (int r = 0; r < 8; ++r) {
        int t = tBase + i * 16 + r;
        float v = acc[i][j][r] * 0.03125f;         // 1/sqrt(1024)
        if (s > t) v = -__builtin_inff();
        S[((size_t)b * T_DIM + t) * T_DIM + s] = v;
      }
    }
}

// ---------------------------------------------------------------------------
// Kernel 3: row softmax; one wave32 per (b,t) row, shuffle reductions.
// ---------------------------------------------------------------------------
__global__ __launch_bounds__(256) void softmax_kernel(
    const float* __restrict__ S, bf16_t* __restrict__ P)
{
  const int rowIdx = blockIdx.x * 8 + (threadIdx.x >> 5);
  const int lane   = threadIdx.x & 31;
  const int t      = rowIdx & (T_DIM - 1);
  const float* srow = S + (size_t)rowIdx * T_DIM;
  bf16_t*      prow = P + (size_t)rowIdx * T_DIM;
  const int n = t + 1;

  float m = -__builtin_inff();
  for (int s = lane; s < n; s += 32) m = fmaxf(m, srow[s]);
#pragma unroll
  for (int off = 16; off; off >>= 1) m = fmaxf(m, __shfl_xor(m, off));

  float sum = 0.0f;
  for (int s = lane; s < n; s += 32) sum += __expf(srow[s] - m);
#pragma unroll
  for (int off = 16; off; off >>= 1) sum += __shfl_xor(sum, off);
  const float inv = 1.0f / sum;

  for (int s = lane; s < T_DIM; s += 32) {
    float v = 0.0f;
    if (s < n) v = __expf(srow[s] - m) * inv;
    prow[s] = (bf16_t)v;
  }
}

// ---------------------------------------------------------------------------
// Kernel 4: out[b,t,c] = sum_s P[b,t,s]*V[b,s,c]; k-loop ends at s<=t0+127.
// ---------------------------------------------------------------------------
__global__ __launch_bounds__(256) void av_kernel(
    const bf16_t* __restrict__ P, const bf16_t* __restrict__ V,
    float* __restrict__ out)
{
  const int b  = blockIdx.z;
  const int t0 = blockIdx.x * 128;
  const int c0 = blockIdx.y * 128;
  const bf16_t* Pp = P + (size_t)b * T_DIM * T_DIM + (size_t)t0 * T_DIM;
  const bf16_t* Vp = V + (size_t)b * T_DIM * C_DIM + c0;

  const int tid  = threadIdx.x;
  const int wave = tid >> 5, wm = wave >> 1, wn = wave & 1;

  v8f acc[2][4];
#pragma unroll
  for (int i = 0; i < 2; ++i)
#pragma unroll
    for (int j = 0; j < 4; ++j) acc[i][j] = zero_v8f();

  const int NK = (t0 + 128) / 32;

#if USE_ASYNC
  __shared__ bf16_t ldsA[2][128 * LDSK];
  __shared__ bf16_t ldsB[2][128 * LDSK];
  bf16x8 rv[2];
  gload_v_tile(Vp, 0, rv);
  async_bf16_tile(Pp, T_DIM, ldsA[0]);
  for (int kt = 0; kt < NK; ++kt) {
    const int cur = kt & 1;
    WAIT_ASYNC();
    store_vT_tile(ldsB[cur], rv);         // ldsB[cur] last read 2 iters ago: safe
    __syncthreads();
    if (kt + 1 < NK) {
      async_bf16_tile(Pp + (kt + 1) * 32, T_DIM, ldsA[cur ^ 1]);
      gload_v_tile(Vp, (kt + 1) * 32, rv);
    }
    mma_tiles(ldsA[cur], ldsB[cur], wm, wn, acc);
  }
#else
  __shared__ bf16_t ldsA[128 * LDSK];
  __shared__ bf16_t ldsB[128 * LDSK];
  bf16x8 ra[2], rv[2];
  gload_bf16_tile(Pp, T_DIM, 0, ra);
  gload_v_tile(Vp, 0, rv);
  for (int kt = 0; kt < NK; ++kt) {
    __syncthreads();
    store_bf16_tile(ldsA, ra);
    store_vT_tile(ldsB, rv);
    __syncthreads();
    if (kt + 1 < NK) {
      gload_bf16_tile(Pp, T_DIM, (kt + 1) * 32, ra);
      gload_v_tile(Vp, (kt + 1) * 32, rv);
    }
    mma_tiles(ldsA, ldsB, wm, wn, acc);
  }
#endif

  const int lane  = tid & 31;
  const int tBase = t0 + wm * 32 + ((lane >> 4) << 3);
  const int cBase = c0 + wn * 64 + (lane & 15);
#pragma unroll
  for (int i = 0; i < 2; ++i)
#pragma unroll
    for (int j = 0; j < 4; ++j) {
      int c = cBase + j * 16;
#pragma unroll
      for (int r = 0; r < 8; ++r)
        out[((size_t)b * T_DIM + tBase + i * 16 + r) * C_DIM + c] = acc[i][j][r];
    }
}

// ---------------------------------------------------------------------------
extern "C" void kernel_launch(void* const* d_in, const int* in_sizes, int n_in,
                              void* d_out, int out_size, void* d_ws, size_t ws_size,
                              hipStream_t stream) {
  const float* x  = (const float*)d_in[0];
  const float* wq = (const float*)d_in[1];
  const float* bq = (const float*)d_in[2];
  const float* wk = (const float*)d_in[3];
  const float* bk = (const float*)d_in[4];
  const float* wv = (const float*)d_in[5];
  const float* bv = (const float*)d_in[6];
  float* out = (float*)d_out;

  // Workspace: Q(32MB) K(32MB) V(32MB) S f32(128MB) P bf16(64MB)
  char* ws = (char*)d_ws;
  bf16_t* Q = (bf16_t*)(ws);
  bf16_t* K = (bf16_t*)(ws + (size_t)33554432);
  bf16_t* V = (bf16_t*)(ws + (size_t)67108864);
  float*  S = (float*) (ws + (size_t)100663296);
  bf16_t* P = (bf16_t*)(ws + (size_t)234881024);

  qkv_kernel    <<<dim3(128, 8, 3),  256, 0, stream>>>(x, wq, bq, wk, bk, wv, bv, Q, K, V);
  scores_kernel <<<dim3(16, 16, 8),  256, 0, stream>>>(K, Q, S);
  softmax_kernel<<<dim3(2048),       256, 0, stream>>>(S, P);
  av_kernel     <<<dim3(16, 8, 8),   256, 0, stream>>>(P, V, out);
}
